// NeRF_66211215835246
// MI455X (gfx1250) — compile-verified, matrix operand order checked by reading, not run on hardware
//
#include <hip/hip_runtime.h>
#include <hip/hip_bf16.h>

typedef __attribute__((ext_vector_type(16))) _Float16 v16h;
typedef __attribute__((ext_vector_type(8)))  _Float16 v8h;
typedef __attribute__((ext_vector_type(8)))  float    v8f;
typedef __attribute__((ext_vector_type(4)))  int      v4i;

#define AST 336            // activation row stride in f16 (>=320, keeps 16B alignment)
#define NEARP 2.0f
#define FARP  6.0f

// weight-fragment region offsets (f16 elements) inside one param set
#define WO0  0            // W0:  K=64  (2 ksteps)  N=256 (16 ntiles)
#define WO1  16384
#define WO2  81920
#define WO3  147456
#define WO4  212992
#define WO5  278528       // W5:  K=320 (10)        N=256
#define WO6  360448
#define WO7  425984
#define WO8  491520       // [wa|wf]: K=256 (8)     N=272 (17 ntiles)
#define WO9  561152       // wv:  K=288 (9)         N=128 (8)
#define WO10 598016       // wr:  K=128 (4)         N=16  (1)
#define WTOT 600064

// packed bias offsets (f32 elements)
#define BO_HEAD 2048
#define BO_WV   2320
#define BO_WR   2448
#define BTOT    2464

// CDNA5 async global->LDS path (guarded: falls back to vector copies).
// Signature (probe-confirmed): (int4 AS1*, int4 AS3*, imm offset, imm cpol).
#if defined(__HIP_DEVICE_COMPILE__) && \
    __has_builtin(__builtin_amdgcn_global_load_async_to_lds_b128) && \
    __has_builtin(__builtin_amdgcn_s_wait_asynccnt)
#define HAVE_ASYNC_LDS 1
#define AS_GLOBAL __attribute__((address_space(1)))
#define AS_LDS    __attribute__((address_space(3)))
#endif

// ---------------------------------------------------------------------------
// Weight repack: f32 row-major [K,N]  ->  f16 WMMA B-fragments.
// Fragment (nt,ks) is 512 f16: lane-major, each lane's 16 f16 are contiguous
// and match the v16h the lane feeds to v_wmma_f32_16x16x32_f16.
// remap=1 implements the NeRF skip concat padding (row63 = zero pad, rows
// 64..319 come from source rows 63..318).
// ---------------------------------------------------------------------------
__global__ void conv_wfrag(const float* __restrict__ src, int Ko, int No,
                           int ksteps, int ntiles, int remap,
                           _Float16* __restrict__ dst)
{
    int idx = blockIdx.x * blockDim.x + threadIdx.x;
    int total = ntiles * ksteps * 512;
    if (idx >= total) return;
    int frag = idx >> 9;
    int e    = idx & 511;
    int lane = e >> 4;
    int j    = e & 15;
    int nt   = frag / ksteps;
    int ks   = frag - nt * ksteps;
    int half = lane >> 4;
    int n    = nt * 16 + (lane & 15);
    int kl   = (j < 8) ? (half * 8 + j) : (16 + half * 8 + (j - 8));
    int k    = ks * 32 + kl;
    int ksrc = k;
    bool ok = true;
    if (remap) { if (k == 63) ok = false; else if (k > 63) ksrc = k - 1; }
    float v = 0.0f;
    if (ok && ksrc < Ko && n < No) v = src[(size_t)ksrc * No + n];
    dst[idx] = (_Float16)v;
}

__global__ void pack_bias(const float* b0, const float* b1, const float* b2, const float* b3,
                          const float* b4, const float* b5, const float* b6, const float* b7,
                          const float* ba, const float* bf, const float* bv, const float* br,
                          float* __restrict__ dst)
{
    int i = blockIdx.x * blockDim.x + threadIdx.x;
    if (i >= BTOT) return;
    float v = 0.0f;
    if (i < 2048) {
        const float* bs[8] = {b0, b1, b2, b3, b4, b5, b6, b7};
        v = bs[i >> 8][i & 255];
    } else if (i < 2064) {
        v = (i == 2048) ? ba[0] : 0.0f;               // alpha bias in padded tile 0
    } else if (i < BO_WV) {
        v = bf[i - 2064];                             // feat bias, cols 16..271
    } else if (i < BO_WR) {
        v = bv[i - BO_WV];
    } else {
        int c = i - BO_WR;
        v = (c < 3) ? br[c] : 0.0f;
    }
    dst[i] = v;
}

// ---------------------------------------------------------------------------
// Positional encodings (f16, padded widths 64 / 32)
// ---------------------------------------------------------------------------
__global__ void posenc_xyz(const float* __restrict__ ro, const float* __restrict__ rd,
                           const float* __restrict__ zin, int S,
                           _Float16* __restrict__ xyz_e)
{
    int s = blockIdx.x * blockDim.x + threadIdx.x;
    if (s >= 1024 * S) return;
    int r = s / S, i = s - r * S;
    float zv = zin ? zin[s] : (NEARP + (FARP - NEARP) * i / (float)(S - 1));
    _Float16* row = xyz_e + (size_t)s * 64;
    float p[3];
#pragma unroll
    for (int c = 0; c < 3; ++c) {
        p[c] = ro[r * 3 + c] + rd[r * 3 + c] * zv;
        row[c] = (_Float16)p[c];
    }
    float f = 1.0f;
    for (int l = 0; l < 10; ++l) {
#pragma unroll
        for (int c = 0; c < 3; ++c) {
            row[3 + l * 6 + c]     = (_Float16)sinf(p[c] * f);
            row[3 + l * 6 + 3 + c] = (_Float16)cosf(p[c] * f);
        }
        f *= 2.0f;
    }
    row[63] = (_Float16)0.0f;
}

__global__ void posenc_dir(const float* __restrict__ rd, _Float16* __restrict__ dir_e)
{
    int r = blockIdx.x * blockDim.x + threadIdx.x;
    if (r >= 1024) return;
    float d[3]; float n2 = 0.0f;
#pragma unroll
    for (int c = 0; c < 3; ++c) { d[c] = rd[r * 3 + c]; n2 += d[c] * d[c]; }
    float inv = rsqrtf(n2);
    _Float16* row = dir_e + (size_t)r * 32;
    float v[3];
#pragma unroll
    for (int c = 0; c < 3; ++c) { v[c] = d[c] * inv; row[c] = (_Float16)v[c]; }
    float f = 1.0f;
    for (int l = 0; l < 4; ++l) {
#pragma unroll
        for (int c = 0; c < 3; ++c) {
            row[3 + l * 6 + c]     = (_Float16)sinf(v[c] * f);
            row[3 + l * 6 + 3 + c] = (_Float16)cosf(v[c] * f);
        }
        f *= 2.0f;
    }
    for (int c = 27; c < 32; ++c) row[c] = (_Float16)0.0f;
}

// ---------------------------------------------------------------------------
// WMMA fragment helpers
// A: 16x32 f16, lanes 0-15 hold row M (K 0..7,16..23), lanes 16-31 (K 8..15,24..31)
// ---------------------------------------------------------------------------
__device__ __forceinline__ v16h ld_a(const _Float16* act, int stride, int mt, int ks, int lane)
{
    const _Float16* p = act + (size_t)(mt * 16 + (lane & 15)) * stride
                        + ks * 32 + ((lane >> 4) << 3);
    v8h c0 = *(const v8h*)(p);
    v8h c1 = *(const v8h*)(p + 16);
    return __builtin_shufflevector(c0, c1, 0,1,2,3,4,5,6,7,8,9,10,11,12,13,14,15);
}

__device__ __forceinline__ v16h ld_b(const _Float16* wf, int fragIdx, int lane)
{
    const _Float16* q = wf + ((size_t)fragIdx << 9) + (lane << 4);
    v8h c0 = *(const v8h*)(q);
    v8h c1 = *(const v8h*)(q + 8);
    return __builtin_shufflevector(c0, c1, 0,1,2,3,4,5,6,7,8,9,10,11,12,13,14,15);
}

__device__ __forceinline__ void prefetch_w(const _Float16* p, int n, int t)
{
    for (int i = t * 64; i < n; i += 256 * 64)       // one 128B line per step
        __builtin_prefetch(p + i, 0, 1);             // -> global_prefetch_b8
}

__device__ __forceinline__ void store_tile(_Float16* actOut, int outOff, int nt, int mt,
                                           int lane, const v8f& acc, float bcol, int relu)
{
    _Float16* o = actOut + (size_t)(mt * 16 + ((lane >> 4) << 3)) * AST
                  + outOff + nt * 16 + (lane & 15);
#pragma unroll
    for (int r2 = 0; r2 < 8; ++r2) {
        float v = acc[r2] + bcol;
        if (relu) v = fmaxf(v, 0.0f);
        o[(size_t)r2 * AST] = (_Float16)v;
    }
}

// One fused layer: LDS activations [128 x K] @ wfrag [K x 16*ntiles] -> LDS out.
// KSTEPS is compile-time so per-N-tile B fragments live in registers across all
// 8 M-tiles; M-tiles are processed in PAIRS (two independent accumulator
// chains) so WMMA->VALU hazard slots are filled with useful work, not v_nop.
template <int KSTEPS>
__device__ __forceinline__ void gemm_ld(const _Float16* actIn, int inStride,
                                        _Float16* actOut, int outOff,
                                        const _Float16* wf, const float* bias,
                                        int ntiles, int relu,
                                        int wid, int lane)
{
    for (int nt = wid; nt < ntiles; nt += 8) {
        float bcol = bias[nt * 16 + (lane & 15)];
        v16h bfr[KSTEPS];
#pragma unroll
        for (int ks = 0; ks < KSTEPS; ++ks)
            bfr[ks] = ld_b(wf, nt * KSTEPS + ks, lane);
        for (int mt = 0; mt < 8; mt += 2) {
            v8f acc0 = {}, acc1 = {};
#pragma unroll
            for (int ks = 0; ks < KSTEPS; ++ks) {
                v16h a0 = ld_a(actIn, inStride, mt,     ks, lane);
                v16h a1 = ld_a(actIn, inStride, mt + 1, ks, lane);
                acc0 = __builtin_amdgcn_wmma_f32_16x16x32_f16(
                        false, a0, false, bfr[ks], (short)0, acc0, false, false);
                acc1 = __builtin_amdgcn_wmma_f32_16x16x32_f16(
                        false, a1, false, bfr[ks], (short)0, acc1, false, false);
            }
            store_tile(actOut, outOff, nt, mt,     lane, acc0, bcol, relu);
            store_tile(actOut, outOff, nt, mt + 1, lane, acc1, bcol, relu);
        }
    }
}

// ---------------------------------------------------------------------------
// Fused NeRF field: 128 samples per block, whole MLP chain resident in LDS.
// ---------------------------------------------------------------------------
__global__ void __launch_bounds__(256) mlp_kernel(
    const _Float16* __restrict__ xyz_e, const _Float16* __restrict__ dir_e,
    const _Float16* __restrict__ wfrag, const float* __restrict__ bias,
    float* __restrict__ alpha_out, float* __restrict__ rgb_out, int S)
{
    extern __shared__ _Float16 smem[];
    _Float16* xyzR = smem;               // 128 x 64  (kept for skip concat)
    _Float16* dirR = smem + 128 * 64;    // 128 x 32
    _Float16* bufA = smem + 128 * 96;    // 128 x AST
    _Float16* bufB = bufA + 128 * AST;   // 128 x AST
    const int t = threadIdx.x;
    const int wid = t >> 5, lane = t & 31;
    const size_t base = (size_t)blockIdx.x * 128;

    // stage encodings into LDS
#if defined(HAVE_ASYNC_LDS)
    {   // CDNA5 async DMA: global -> LDS without VGPR round-trip (ASYNCcnt)
        for (int i = t; i < 1024; i += 256)
            __builtin_amdgcn_global_load_async_to_lds_b128(
                (AS_GLOBAL v4i*)(xyz_e + base * 64 + (size_t)i * 8),
                (AS_LDS v4i*)(xyzR + (size_t)i * 8), 0, 0);
        for (int i = t; i < 512; i += 256) {
            int row = i >> 2, c = i & 3;
            size_t ray = (base + (size_t)row) / (size_t)S;
            __builtin_amdgcn_global_load_async_to_lds_b128(
                (AS_GLOBAL v4i*)(dir_e + ray * 32 + (size_t)c * 8),
                (AS_LDS v4i*)(dirR + (size_t)i * 8), 0, 0);
        }
        __builtin_amdgcn_s_wait_asynccnt(0);
    }
#else
    {
        const v8h* src = (const v8h*)(xyz_e + base * 64);
        v8h* dst = (v8h*)xyzR;
        for (int i = t; i < 1024; i += 256) dst[i] = src[i];
        for (int i = t; i < 512; i += 256) {
            int row = i >> 2, c = i & 3;
            size_t ray = (base + (size_t)row) / (size_t)S;
            ((v8h*)dirR)[i] = ((const v8h*)(dir_e + ray * 32))[c];
        }
    }
#endif
    __syncthreads();

    prefetch_w(wfrag + WO1, 65536, t);
    gemm_ld<2>(xyzR, 64, bufA, 0,  wfrag + WO0, bias + 0,    16, 1, wid, lane);
    __syncthreads();
    prefetch_w(wfrag + WO2, 65536, t);
    gemm_ld<8>(bufA, AST, bufB, 0, wfrag + WO1, bias + 256,  16, 1, wid, lane);
    __syncthreads();
    prefetch_w(wfrag + WO3, 65536, t);
    gemm_ld<8>(bufB, AST, bufA, 0, wfrag + WO2, bias + 512,  16, 1, wid, lane);
    __syncthreads();
    prefetch_w(wfrag + WO4, 65536, t);
    gemm_ld<8>(bufA, AST, bufB, 0, wfrag + WO3, bias + 768,  16, 1, wid, lane);
    __syncthreads();
    prefetch_w(wfrag + WO5, 81920, t);
    gemm_ld<8>(bufB, AST, bufA, 64, wfrag + WO4, bias + 1024, 16, 1, wid, lane);
    __syncthreads();
    // skip connection: xyz encodings into cols 0..63 of bufA
    for (int i = t; i < 1024; i += 256) {
        int row = i >> 3, c = i & 7;
        *(v8h*)(bufA + (size_t)row * AST + c * 8) = ((const v8h*)xyzR)[i];
    }
    __syncthreads();
    prefetch_w(wfrag + WO6, 65536, t);
    gemm_ld<10>(bufA, AST, bufB, 0, wfrag + WO5, bias + 1280, 16, 1, wid, lane);
    __syncthreads();
    prefetch_w(wfrag + WO7, 65536, t);
    gemm_ld<8>(bufB, AST, bufA, 0, wfrag + WO6, bias + 1536, 16, 1, wid, lane);
    __syncthreads();
    prefetch_w(wfrag + WO8, 69632, t);
    gemm_ld<8>(bufA, AST, bufB, 0, wfrag + WO7, bias + 1792, 16, 1, wid, lane);
    __syncthreads();
    // head: [alpha(16-pad) | feat(256)] = h @ [wa | wf]  (linear)
    prefetch_w(wfrag + WO9, 36864, t);
    gemm_ld<8>(bufB, AST, bufA, 0, wfrag + WO8, bias + BO_HEAD, 17, 0, wid, lane);
    __syncthreads();
    if (t < 128) alpha_out[base + t] = (float)bufA[(size_t)t * AST];
    // append dir encodings at cols 272..303 (feat occupies 16..271)
    for (int i = t; i < 512; i += 256) {
        int row = i >> 2, c = i & 3;
        *(v8h*)(bufA + (size_t)row * AST + 272 + c * 8) = ((const v8h*)dirR)[i];
    }
    __syncthreads();
    prefetch_w(wfrag + WO10, 2048, t);
    gemm_ld<9>(bufA + 16, AST, bufB, 0, wfrag + WO9, bias + BO_WV, 8, 1, wid, lane);
    __syncthreads();
    // rgb head: single N-tile, wave 0 only (EXEC all-ones inside the wave)
    if (wid == 0) {
        float bcol = bias[BO_WR + (lane & 15)];
        v16h bfr[4];
#pragma unroll
        for (int ks = 0; ks < 4; ++ks) bfr[ks] = ld_b(wfrag + WO10, ks, lane);
        for (int mt = 0; mt < 8; mt += 2) {
            v8f acc0 = {}, acc1 = {};
#pragma unroll
            for (int ks = 0; ks < 4; ++ks) {
                v16h a0 = ld_a(bufB, AST, mt,     ks, lane);
                v16h a1 = ld_a(bufB, AST, mt + 1, ks, lane);
                acc0 = __builtin_amdgcn_wmma_f32_16x16x32_f16(
                        false, a0, false, bfr[ks], (short)0, acc0, false, false);
                acc1 = __builtin_amdgcn_wmma_f32_16x16x32_f16(
                        false, a1, false, bfr[ks], (short)0, acc1, false, false);
            }
            int col = lane & 15;
            if (col < 3) {
#pragma unroll
                for (int r2 = 0; r2 < 8; ++r2) {
                    int row0 = mt * 16 + ((lane >> 4) << 3) + r2;
                    float v0 = acc0[r2] + bcol;
                    rgb_out[(base + row0) * 3 + col] = 1.0f / (1.0f + expf(-v0));
                    float v1 = acc1[r2] + bcol;
                    rgb_out[(base + row0 + 16) * 3 + col] = 1.0f / (1.0f + expf(-v1));
                }
            }
        }
    }
}

// ---------------------------------------------------------------------------
// Volume rendering + deterministic importance sampling (1 thread / ray)
// ---------------------------------------------------------------------------
__global__ void render_kernel(const float* __restrict__ alpha_raw, const float* __restrict__ rgbv,
                              const float* __restrict__ rays_d, const float* __restrict__ zin,
                              int S, float* __restrict__ out, int colOff,
                              float* __restrict__ zfine_out)
{
    int r = blockIdx.x * blockDim.x + threadIdx.x;
    if (r >= 1024) return;
    float z[128], w[128];
    for (int i = 0; i < S; ++i)
        z[i] = zin ? zin[(size_t)r * S + i]
                   : (NEARP + (FARP - NEARP) * i / (float)(S - 1));
    float dx = rays_d[r * 3], dy = rays_d[r * 3 + 1], dz = rays_d[r * 3 + 2];
    float dn = sqrtf(dx * dx + dy * dy + dz * dz);
    float T = 1.0f, cr = 0.0f, cg = 0.0f, cb = 0.0f;
    for (int i = 0; i < S; ++i) {
        float dist = ((i < S - 1) ? (z[i + 1] - z[i]) : 1e10f) * dn;
        float raw = alpha_raw[(size_t)r * S + i];
        float al = 1.0f - expf(-fmaxf(raw, 0.0f) * dist);
        float wi = al * T;
        T *= (1.0f - al + 1e-10f);
        w[i] = wi;
        const float* c = rgbv + ((size_t)r * S + i) * 3;
        cr += wi * c[0]; cg += wi * c[1]; cb += wi * c[2];
    }
    out[r * 6 + colOff + 0] = cr;
    out[r * 6 + colOff + 1] = cg;
    out[r * 6 + colOff + 2] = cb;

    if (zfine_out) {   // coarse pass only: sample_pdf (det) + sorted merge
        float bins[63], cdf[63];
        for (int i = 0; i < 63; ++i) bins[i] = 0.5f * (z[i] + z[i + 1]);
        float sum = 0.0f;
        for (int i = 0; i < 62; ++i) sum += w[i + 1] + 1e-5f;
        cdf[0] = 0.0f;
        float acc = 0.0f;
        for (int i = 0; i < 62; ++i) { acc += (w[i + 1] + 1e-5f) / sum; cdf[i + 1] = acc; }
        float zs[64];
        for (int j = 0; j < 64; ++j) {
            float u = (float)j / 63.0f;
            int ind = 63;
            for (int i = 0; i < 63; ++i) { if (cdf[i] > u) { ind = i; break; } }
            int below = (ind - 1 > 0) ? ind - 1 : 0;
            int above = (ind < 62) ? ind : 62;
            float c0 = cdf[below], c1 = cdf[above];
            float b0 = bins[below], b1 = bins[above];
            float den = (c1 - c0 < 1e-5f) ? 1.0f : (c1 - c0);
            zs[j] = b0 + (u - c0) / den * (b1 - b0);
        }
        int ai = 0, bi = 0;
        for (int k = 0; k < 128; ++k) {
            float v;
            if (ai < 64 && (bi >= 64 || z[ai] <= zs[bi])) v = z[ai++];
            else v = zs[bi++];
            zfine_out[(size_t)r * 128 + k] = v;
        }
    }
}

// ---------------------------------------------------------------------------
extern "C" void kernel_launch(void* const* d_in, const int* in_sizes, int n_in,
                              void* d_out, int out_size, void* d_ws, size_t ws_size,
                              hipStream_t stream)
{
    (void)in_sizes; (void)n_in; (void)out_size; (void)ws_size;
    const float* rays_o = (const float*)d_in[0];
    const float* rays_d = (const float*)d_in[1];

    char* ws = (char*)d_ws;
    size_t off = 0;
    auto carve = [&](size_t bytes) {
        char* p = ws + off;
        off += (bytes + 255) & ~(size_t)255;
        return p;
    };

    _Float16* wfrag[2]; float* bias[2];
    wfrag[0] = (_Float16*)carve((size_t)WTOT * 2);
    wfrag[1] = (_Float16*)carve((size_t)WTOT * 2);
    bias[0]  = (float*)carve((size_t)BTOT * 4);
    bias[1]  = (float*)carve((size_t)BTOT * 4);
    _Float16* dirE = (_Float16*)carve((size_t)1024 * 32 * 2);
    _Float16* xyzE = (_Float16*)carve((size_t)131072 * 64 * 2);  // shared coarse/fine
    float* alphaC = (float*)carve((size_t)65536 * 4);
    float* rgbC   = (float*)carve((size_t)65536 * 3 * 4);
    float* alphaF = (float*)carve((size_t)131072 * 4);
    float* rgbF   = (float*)carve((size_t)131072 * 3 * 4);
    float* zfine  = (float*)carve((size_t)1024 * 128 * 4);

    auto conv = [&](const float* src, int Ko, int No, int ksteps, int ntiles,
                    int remap, _Float16* dst) {
        int total = ntiles * ksteps * 512;
        conv_wfrag<<<(total + 255) / 256, 256, 0, stream>>>(src, Ko, No, ksteps, ntiles, remap, dst);
    };

    static const int WOL[8] = {WO0, WO1, WO2, WO3, WO4, WO5, WO6, WO7};
    for (int s = 0; s < 2; ++s) {
        int pi = 2 + s * 24;
        const float *Wm[8], *Bm[8];
        for (int i = 0; i < 8; ++i) {
            Wm[i] = (const float*)d_in[pi + 2 * i];
            Bm[i] = (const float*)d_in[pi + 2 * i + 1];
        }
        const float* wa = (const float*)d_in[pi + 16]; const float* ba = (const float*)d_in[pi + 17];
        const float* wff= (const float*)d_in[pi + 18]; const float* bf = (const float*)d_in[pi + 19];
        const float* wv = (const float*)d_in[pi + 20]; const float* bv = (const float*)d_in[pi + 21];
        const float* wr = (const float*)d_in[pi + 22]; const float* br = (const float*)d_in[pi + 23];

        conv(Wm[0], 63, 256, 2, 16, 0, wfrag[s] + WO0);
        for (int i = 1; i < 8; ++i) {
            if (i == 5) conv(Wm[5], 319, 256, 10, 16, 1, wfrag[s] + WO5);
            else        conv(Wm[i], 256, 256, 8,  16, 0, wfrag[s] + WOL[i]);
        }
        conv(wa,  256, 1,   8, 1,  0, wfrag[s] + WO8);
        conv(wff, 256, 256, 8, 16, 0, wfrag[s] + WO8 + 8 * 512);
        conv(wv,  283, 128, 9, 8,  0, wfrag[s] + WO9);
        conv(wr,  128, 3,   4, 1,  0, wfrag[s] + WO10);
        pack_bias<<<(BTOT + 255) / 256, 256, 0, stream>>>(
            Bm[0], Bm[1], Bm[2], Bm[3], Bm[4], Bm[5], Bm[6], Bm[7],
            ba, bf, bv, br, bias[s]);
    }

    posenc_dir<<<4, 256, 0, stream>>>(rays_d, dirE);

    const size_t smemBytes = (size_t)128 * (64 + 32 + 2 * AST) * 2;   // ~192 KB / WGP

    // coarse pass
    posenc_xyz<<<256, 256, 0, stream>>>(rays_o, rays_d, nullptr, 64, xyzE);
    mlp_kernel<<<512, 256, smemBytes, stream>>>(xyzE, dirE, wfrag[0], bias[0], alphaC, rgbC, 64);
    render_kernel<<<4, 256, 0, stream>>>(alphaC, rgbC, rays_d, nullptr, 64,
                                         (float*)d_out, 0, zfine);

    // fine pass (64 stratified + 64 importance samples, merged & sorted)
    posenc_xyz<<<512, 256, 0, stream>>>(rays_o, rays_d, zfine, 128, xyzE);
    mlp_kernel<<<1024, 256, smemBytes, stream>>>(xyzE, dirE, wfrag[1], bias[1], alphaF, rgbF, 128);
    render_kernel<<<4, 256, 0, stream>>>(alphaF, rgbF, rays_d, zfine, 128,
                                         (float*)d_out, 3, nullptr);
}
